// IntrinsicTension_17540646437627
// MI455X (gfx1250) — matrix operand order repaired and checked by reading, not compiled, not run on hardware
//
#include <hip/hip_runtime.h>
#include <hip/hip_bf16.h>

typedef __attribute__((ext_vector_type(16))) _Float16 v16h;
typedef __attribute__((ext_vector_type(8)))  float    v8f;
typedef __attribute__((ext_vector_type(4)))  float    f4v;

__device__ __forceinline__ float gelu_exact(float x) {
    return 0.5f * x * (1.0f + erff(x * 0.70710678118654752f));
}

// ---------------------------------------------------------------------------
// Phase 1: fused streaming reduction over D=131072.
// 8 accumulators: 0=|d|^2 1=|v|^2 2=|fs|^2 3=|ft|^2 4=v.d 5=fs.ft 6=v.fs 7=v.ft
// Grid: 256 rows x 4 splits = 1024 blocks, 256 threads. NT 128-bit loads.
// ---------------------------------------------------------------------------
__global__ __launch_bounds__(256) void it_partials(
    const float* __restrict__ x,  const float* __restrict__ xr,
    const float* __restrict__ fs, const float* __restrict__ ft,
    const float* __restrict__ v,  float* __restrict__ part) {
    constexpr int D = 131072, CH = 32768;
    const int row = blockIdx.x >> 2, sp = blockIdx.x & 3;
    const size_t base4 = (((size_t)row * D) + (size_t)sp * CH) >> 2;
    const f4v* px  = (const f4v*)x  + base4;
    const f4v* pxr = (const f4v*)xr + base4;
    const f4v* pfs = (const f4v*)fs + base4;
    const f4v* pft = (const f4v*)ft + base4;
    const f4v* pv  = (const f4v*)v  + base4;

    float acc[8] = {0.f,0.f,0.f,0.f,0.f,0.f,0.f,0.f};
    for (int i = threadIdx.x; i < CH / 4; i += 256) {
        f4v a = __builtin_nontemporal_load(px  + i);
        f4v b = __builtin_nontemporal_load(pxr + i);
        f4v e = __builtin_nontemporal_load(pfs + i);
        f4v f = __builtin_nontemporal_load(pft + i);
        f4v c = __builtin_nontemporal_load(pv  + i);
        #pragma unroll
        for (int q = 0; q < 4; ++q) {
            float dv = b[q] - a[q];
            float vv = c[q], sv = e[q], tv = f[q];
            acc[0] += dv * dv;
            acc[1] += vv * vv;
            acc[2] += sv * sv;
            acc[3] += tv * tv;
            acc[4] += vv * dv;
            acc[5] += sv * tv;
            acc[6] += vv * sv;
            acc[7] += vv * tv;
        }
    }

    __shared__ float sred[256][9];
    #pragma unroll
    for (int k = 0; k < 8; ++k) sred[threadIdx.x][k] = acc[k];
    __syncthreads();
    for (int s = 128; s > 0; s >>= 1) {
        if (threadIdx.x < s) {
            #pragma unroll
            for (int k = 0; k < 8; ++k)
                sred[threadIdx.x][k] += sred[threadIdx.x + s][k];
        }
        __syncthreads();
    }
    if (threadIdx.x == 0) {
        #pragma unroll
        for (int k = 0; k < 8; ++k)
            part[(size_t)blockIdx.x * 8 + k] = sred[0][k];
    }
}

// ---------------------------------------------------------------------------
// Phase 1b: combine 4 splits per row, compute the 8 invariants, scale.
// ---------------------------------------------------------------------------
__global__ __launch_bounds__(256) void it_finalize(
    const float* __restrict__ part, const float* __restrict__ inv_scales,
    float* __restrict__ sinv) {
    const int r = threadIdx.x;  // 256 rows, 1 block
    float s[8] = {0.f,0.f,0.f,0.f,0.f,0.f,0.f,0.f};
    for (int sp = 0; sp < 4; ++sp) {
        #pragma unroll
        for (int k = 0; k < 8; ++k)
            s[k] += part[(size_t)(r * 4 + sp) * 8 + k];
    }
    const float eps = 1e-8f;
    float dn = sqrtf(s[0]) + eps;
    float vn = sqrtf(s[1]) + eps;
    float sn = sqrtf(s[2]) + eps;
    float tn = sqrtf(s[3]) + eps;
    float inv[8];
    inv[0] = log1pf(dn);
    inv[1] = log1pf(vn);
    inv[2] = s[4] / (vn * dn);
    inv[3] = s[5] / (sn * tn);
    inv[4] = s[6] / (vn * sn);
    inv[5] = s[7] / (vn * tn);
    inv[6] = log1pf(vn * vn + dn * dn);
    inv[7] = logf(tn / sn);
    #pragma unroll
    for (int k = 0; k < 8; ++k) sinv[r * 8 + k] = inv[k] * inv_scales[k];
}

// ---------------------------------------------------------------------------
// Phase 2: MLP head. One block = 16 batch rows, 256 threads = 8 waves.
// Layers 2/3 via v_wmma_f32_16x16x32_f16 with hi/lo split-precision operands
// (Ah*Bh + Al*Bh + Ah*Bl) accumulated in f32 -> ~fp32 accuracy.
// ---------------------------------------------------------------------------
__global__ __launch_bounds__(256) void it_mlp(
    const float* __restrict__ sinv,
    const float* __restrict__ W1, const float* __restrict__ b1,
    const float* __restrict__ lnw, const float* __restrict__ lnb,
    const float* __restrict__ W2, const float* __restrict__ b2,
    const float* __restrict__ W3, const float* __restrict__ b3,
    const float* __restrict__ W4, const float* __restrict__ b4p,
    const float* __restrict__ eqp, const float* __restrict__ gainp,
    float* __restrict__ out) {
    __shared__ float z[16][128];
    __shared__ float red[16][17][2];
    __shared__ float mu[16], rs[16];
    __shared__ _Float16 a1h[16][136], a1l[16][136];  // layer-2 A tile (hi/lo)
    __shared__ _Float16 a2h[16][136], a2l[16][136];  // layer-3 A tile (hi/lo)
    __shared__ float h3[16][72];
    __shared__ float p4[16][8];

    const int tid = threadIdx.x;
    const int rlocal = tid >> 4;   // 0..15 (batch row within tile)
    const int cseg   = tid & 15;   // 8-column segment
    const int rowbase = blockIdx.x * 16;

    // ---- Layer 1 (K=8, VALU fp32) + running LN stats ----
    float sv[8];
    #pragma unroll
    for (int k = 0; k < 8; ++k) sv[k] = sinv[(rowbase + rlocal) * 8 + k];
    float lsum = 0.f, lsq = 0.f;
    #pragma unroll
    for (int j = 0; j < 8; ++j) {
        int c = cseg * 8 + j;
        float a = b1[c];
        #pragma unroll
        for (int k = 0; k < 8; ++k) a += sv[k] * W1[k * 128 + c];
        z[rlocal][c] = a;
        lsum += a; lsq += a * a;
    }
    red[rlocal][cseg][0] = lsum;
    red[rlocal][cseg][1] = lsq;
    __syncthreads();
    if (tid < 16) {
        float s0 = 0.f, s1 = 0.f;
        for (int i = 0; i < 16; ++i) { s0 += red[tid][i][0]; s1 += red[tid][i][1]; }
        float m = s0 * (1.f / 128.f);
        float var = s1 * (1.f / 128.f) - m * m;
        mu[tid] = m;
        rs[tid] = rsqrtf(var + 1e-5f);
    }
    __syncthreads();
    {   // LN + GELU, emit hi/lo f16 A-tile
        float m = mu[rlocal], r = rs[rlocal];
        #pragma unroll
        for (int j = 0; j < 8; ++j) {
            int c = cseg * 8 + j;
            float hn = (z[rlocal][c] - m) * r * lnw[c] + lnb[c];
            float g = gelu_exact(hn);
            _Float16 hi = (_Float16)g;
            a1h[rlocal][c] = hi;
            a1l[rlocal][c] = (_Float16)(g - (float)hi);
        }
    }
    __syncthreads();

    const int lane = tid & 31, wave = tid >> 5;
    const int m16 = lane & 15, g16 = lane >> 4;

    // ---- Layer 2: h2[16,128] = gelu(h1 @ W2 + b2), 8 waves x 16 cols ----
    {
        const int n0 = wave * 16;
        v8f acc = {};
        #pragma unroll
        for (int kb = 0; kb < 4; ++kb) {
            const int k0 = kb * 32;
            v16h Ah, Al, Bh, Bl;
            #pragma unroll
            for (int h = 0; h < 8; ++h) {
                // A layout: lane -> M=lane%16; halves 0-7: K=k0+8g+h, 8-15: K=k0+16+8g+h
                Ah[h]     = a1h[m16][k0 + 8 * g16 + h];
                Ah[h + 8] = a1h[m16][k0 + 16 + 8 * g16 + h];
                Al[h]     = a1l[m16][k0 + 8 * g16 + h];
                Al[h + 8] = a1l[m16][k0 + 16 + 8 * g16 + h];
            }
            // B layout: lane -> N=n0+lane%16; halves 0-7: K=k0+8g+h, 8-15: K=k0+16+8g+h
            const float* wp = W2 + (size_t)k0 * 128 + n0 + m16;
            #pragma unroll
            for (int h = 0; h < 16; ++h) {
                float wv = wp[(size_t)((h < 8) ? (8 * g16 + h) : (16 + 8 * g16 + h - 8)) * 128];
                _Float16 hi = (_Float16)wv;
                Bh[h] = hi;
                Bl[h] = (_Float16)(wv - (float)hi);
            }
            acc = __builtin_amdgcn_wmma_f32_16x16x32_f16(false, Ah, false, Bh, (short)0, acc, false, false);
            acc = __builtin_amdgcn_wmma_f32_16x16x32_f16(false, Al, false, Bh, (short)0, acc, false, false);
            acc = __builtin_amdgcn_wmma_f32_16x16x32_f16(false, Ah, false, Bl, (short)0, acc, false, false);
        }
        // C layout: lane -> N=n0+lane%16; vgpr j -> M=j+8*(lane/16)
        #pragma unroll
        for (int j = 0; j < 8; ++j) {
            int M = j + 8 * g16, N = n0 + m16;
            float gv = gelu_exact(acc[j] + b2[N]);
            _Float16 hi = (_Float16)gv;
            a2h[M][N] = hi;
            a2l[M][N] = (_Float16)(gv - (float)hi);
        }
    }
    __syncthreads();

    // ---- Layer 3: h3[16,64] = gelu(h2 @ W3 + b3), waves 0..3 ----
    if (wave < 4) {
        const int n0 = wave * 16;
        v8f acc = {};
        #pragma unroll
        for (int kb = 0; kb < 4; ++kb) {
            const int k0 = kb * 32;
            v16h Ah, Al, Bh, Bl;
            #pragma unroll
            for (int h = 0; h < 8; ++h) {
                Ah[h]     = a2h[m16][k0 + 8 * g16 + h];
                Ah[h + 8] = a2h[m16][k0 + 16 + 8 * g16 + h];
                Al[h]     = a2l[m16][k0 + 8 * g16 + h];
                Al[h + 8] = a2l[m16][k0 + 16 + 8 * g16 + h];
            }
            const float* wp = W3 + (size_t)k0 * 64 + n0 + m16;
            #pragma unroll
            for (int h = 0; h < 16; ++h) {
                float wv = wp[(size_t)((h < 8) ? (8 * g16 + h) : (16 + 8 * g16 + h - 8)) * 64];
                _Float16 hi = (_Float16)wv;
                Bh[h] = hi;
                Bl[h] = (_Float16)(wv - (float)hi);
            }
            acc = __builtin_amdgcn_wmma_f32_16x16x32_f16(false, Ah, false, Bh, (short)0, acc, false, false);
            acc = __builtin_amdgcn_wmma_f32_16x16x32_f16(false, Al, false, Bh, (short)0, acc, false, false);
            acc = __builtin_amdgcn_wmma_f32_16x16x32_f16(false, Ah, false, Bl, (short)0, acc, false, false);
        }
        #pragma unroll
        for (int j = 0; j < 8; ++j) {
            int M = j + 8 * g16, N = n0 + m16;
            h3[M][N] = gelu_exact(acc[j] + b3[N]);
        }
    }
    __syncthreads();

    // ---- Layer 4: out = sigmoid(gain * (h3 @ W4 + b4 + eq)) ----
    if (tid < 128) {
        int r = tid >> 3, i = tid & 7;
        float p = 0.f;
        #pragma unroll
        for (int q = 0; q < 8; ++q) p += h3[r][i * 8 + q] * W4[i * 8 + q];
        p4[r][i] = p;
    }
    __syncthreads();
    if (tid < 16) {
        float raw = b4p[0] + eqp[0];
        #pragma unroll
        for (int i = 0; i < 8; ++i) raw += p4[tid][i];
        float gn = gainp[0];
        out[rowbase + tid] = 1.f / (1.f + expf(-gn * raw));
    }
}

extern "C" void kernel_launch(void* const* d_in, const int* in_sizes, int n_in,
                              void* d_out, int out_size, void* d_ws, size_t ws_size,
                              hipStream_t stream) {
    (void)in_sizes; (void)n_in; (void)out_size; (void)ws_size;
    const float* x    = (const float*)d_in[0];
    const float* xr   = (const float*)d_in[1];
    const float* fs   = (const float*)d_in[2];
    const float* ft   = (const float*)d_in[3];
    const float* v    = (const float*)d_in[4];
    const float* invs = (const float*)d_in[5];
    const float* eq   = (const float*)d_in[6];
    const float* gain = (const float*)d_in[7];
    const float* W1   = (const float*)d_in[8];
    const float* b1   = (const float*)d_in[9];
    const float* lnw  = (const float*)d_in[10];
    const float* lnb  = (const float*)d_in[11];
    const float* W2   = (const float*)d_in[12];
    const float* b2   = (const float*)d_in[13];
    const float* W3   = (const float*)d_in[14];
    const float* b3   = (const float*)d_in[15];
    const float* W4   = (const float*)d_in[16];
    const float* b4   = (const float*)d_in[17];
    float* out = (float*)d_out;
    float* wsf = (float*)d_ws;
    float* part = wsf;           // 1024 blocks * 8 floats = 32 KB
    float* sinv = wsf + 8192;    // 256 * 8 floats

    it_partials<<<1024, 256, 0, stream>>>(x, xr, fs, ft, v, part);
    it_finalize<<<1, 256, 0, stream>>>(part, invs, sinv);
    it_mlp<<<16, 256, 0, stream>>>(sinv, W1, b1, lnw, lnb, W2, b2, W3, b3,
                                   W4, b4, eq, gain, out);
}
